// StackGRU_24515673325788
// MI455X (gfx1250) — matrix operand order repaired and checked by reading, not compiled, not run on hardware
//
#include <hip/hip_runtime.h>

#define B_   4096
#define T_   7
#define IN_  512
#define H_   1024
#define H3_  3072
#define OUT_ 512

typedef _Float16 h16;
typedef __attribute__((ext_vector_type(16))) _Float16 v16h;
typedef __attribute__((ext_vector_type(8)))  _Float16 v8h;
typedef __attribute__((ext_vector_type(8)))  float    v8f;

static __device__ __forceinline__ v16h cat8(v8h lo, v8h hi) {
  return __builtin_shufflevector(lo, hi, 0,1,2,3,4,5,6,7,8,9,10,11,12,13,14,15);
}

// Async global->LDS copy of 16 bytes (GLOBAL_LOAD_ASYNC_TO_LDS_B128, GV mode).
// lds_addr: LDS byte address in a VGPR; gaddr: 64-bit global address VGPR pair.
static __device__ __forceinline__ void async_g2l_b128(unsigned lds_addr,
                                                      const void* gaddr) {
  asm volatile("global_load_async_to_lds_b128 %0, %1, off"
               :: "v"(lds_addr), "v"(gaddr) : "memory");
}

// ---------------- elementwise helpers ----------------

__global__ __launch_bounds__(256)
void k_cvt_f16(const float* __restrict__ src, h16* __restrict__ dst, int n) {
  int i = blockIdx.x * blockDim.x + threadIdx.x;
  if (i < n) dst[i] = (h16)src[i];
}

__global__ __launch_bounds__(256)
void k_zero_f32(float* __restrict__ p, int n) {
  int i = blockIdx.x * blockDim.x + threadIdx.x;
  if (i < n) p[i] = 0.0f;
}

__global__ __launch_bounds__(256)
void k_zero_f16(h16* __restrict__ p, int n) {
  int i = blockIdx.x * blockDim.x + threadIdx.x;
  if (i < n) p[i] = (h16)0.0f;
}

// ---------------- WMMA GEMM:  C[M,N] = A[M,K](f16) * B[N,K]^T(f16) + bias[N] ----------------
// Block = 256 threads = 8 waves; C tile 128x128; each wave does 64x32 = 4x2 WMMA tiles.
// K-panels of 32 are staged in double-buffered LDS via async global->LDS copies.
// Launch grid: (N/128, M/128).

#define LDP 40   // padded LDS row pitch in halves (80 B = 20 banks -> conflict-free b128 reads)

__global__ __launch_bounds__(256)
void k_gemm_wmma(const h16* __restrict__ A, int lda,
                 const h16* __restrict__ Bm, int ldb,
                 const float* __restrict__ bias,
                 float* __restrict__ C, int ldc, int K) {
  __shared__ h16 smA[2][128 * LDP];
  __shared__ h16 smB[2][128 * LDP];

  const int tid  = threadIdx.x;
  const int lane = tid & 31;
  const int wave = tid >> 5;
  const int wm   = wave >> 2;                  // 0..1
  const int wn   = wave & 3;                   // 0..3
  const int m_blk = blockIdx.y * 128;
  const int n_blk = blockIdx.x * 128;
  const int hsel = lane >> 4;                  // 0/1 half-wave select
  const int l16  = lane & 15;

  // ---- staging addresses: each thread copies 2 rows x 16B for A and for B per k-step
  const int srow = tid >> 2;                   // 0..63
  const int part = tid & 3;                    // 0..3 (16B chunk within 64B row)
  const h16* gA0 = A  + (size_t)(m_blk + srow) * lda + part * 8;
  const h16* gA1 = gA0 + (size_t)64 * lda;
  const h16* gB0 = Bm + (size_t)(n_blk + srow) * ldb + part * 8;
  const h16* gB1 = gB0 + (size_t)64 * ldb;
  const unsigned lA0 = (unsigned)(size_t)&smA[0][srow * LDP + part * 8];
  const unsigned lA1 = (unsigned)(size_t)&smA[0][(srow + 64) * LDP + part * 8];
  const unsigned lB0 = (unsigned)(size_t)&smB[0][srow * LDP + part * 8];
  const unsigned lB1 = (unsigned)(size_t)&smB[0][(srow + 64) * LDP + part * 8];
  const unsigned bufBytes = (unsigned)(128 * LDP * sizeof(h16));

  // ---- per-wave fragment read offsets (in halves) within a staged tile
  // A frag (16-bit A 16x32 layout): lane<16 K {0..7,16..23}; lanes 16..31 K {8..15,24..31}
  int aro[4];
#pragma unroll
  for (int i = 0; i < 4; ++i) aro[i] = (wm * 64 + i * 16 + l16) * LDP + hsel * 8;
  // B frag (16-bit B 32x16 layout): lane<16 K {0..7,16..23}; lanes 16..31 K {8..15,24..31}
  int bro[2];
#pragma unroll
  for (int j = 0; j < 2; ++j) bro[j] = (wn * 32 + j * 16 + l16) * LDP + hsel * 8;

  const v8f vz = {0.f,0.f,0.f,0.f,0.f,0.f,0.f,0.f};
  v8f acc[4][2];
#pragma unroll
  for (int i = 0; i < 4; ++i)
#pragma unroll
    for (int j = 0; j < 2; ++j) acc[i][j] = vz;

  // ---- prologue: stage first k-panel into buffer 0
  async_g2l_b128(lA0, gA0);
  async_g2l_b128(lA1, gA1);
  async_g2l_b128(lB0, gB0);
  async_g2l_b128(lB1, gB1);

  int buf = 0;
  for (int kk = 0; kk < K; kk += 32) {
    const int nxt = kk + 32;
    if (nxt < K) {
      const unsigned o = (buf ^ 1) * bufBytes;
      async_g2l_b128(lA0 + o, gA0 + nxt);
      async_g2l_b128(lA1 + o, gA1 + nxt);
      async_g2l_b128(lB0 + o, gB0 + nxt);
      async_g2l_b128(lB1 + o, gB1 + nxt);
      asm volatile("s_wait_asynccnt 0x4" ::: "memory");   // current panel landed
    } else {
      asm volatile("s_wait_asynccnt 0x0" ::: "memory");
    }
    __syncthreads();                                      // panel visible to all waves

    const h16* sA = smA[buf];
    const h16* sB = smB[buf];
    v16h a[4], b[2];
#pragma unroll
    for (int i = 0; i < 4; ++i) {
      v8h lo = *(const v8h*)(sA + aro[i]);
      v8h hi = *(const v8h*)(sA + aro[i] + 16);
      a[i] = cat8(lo, hi);
    }
#pragma unroll
    for (int j = 0; j < 2; ++j) {
      v8h lo = *(const v8h*)(sB + bro[j]);
      v8h hi = *(const v8h*)(sB + bro[j] + 16);
      b[j] = cat8(lo, hi);
    }
#pragma unroll
    for (int i = 0; i < 4; ++i)
#pragma unroll
      for (int j = 0; j < 2; ++j)
        acc[i][j] = __builtin_amdgcn_wmma_f32_16x16x32_f16(
            false, a[i], false, b[j], (short)0, acc[i][j], false, false);

    __syncthreads();       // all reads done before next iter overwrites this buffer
    buf ^= 1;
  }

  // Epilogue: C/D layout -> VGPR r, lanes 0..15: M=r, lanes 16..31: M=8+r, N=l16.
#pragma unroll
  for (int j = 0; j < 2; ++j) {
    const int col = n_blk + wn * 32 + j * 16 + l16;
    const float bv = bias ? bias[col] : 0.0f;
#pragma unroll
    for (int i = 0; i < 4; ++i) {
      const int row0 = m_blk + wm * 64 + i * 16 + hsel * 8;
#pragma unroll
      for (int r = 0; r < 8; ++r)
        C[(size_t)(row0 + r) * ldc + col] = acc[i][j][r] + bv;
    }
  }
}

// ---------------- fused GRU gate math ----------------
// GI/GH are [B, 3H] with biases already folded in; gate order (r, z, n).

__global__ __launch_bounds__(256)
void k_gru_gates(const float* __restrict__ GI, const float* __restrict__ GH,
                 const float* __restrict__ hprev,
                 float* __restrict__ hnew, h16* __restrict__ hnew16) {
  int i = blockIdx.x * blockDim.x + threadIdx.x;   // over B*H
  if (i >= B_ * H_) return;
  int b = i >> 10;          // H_ = 1024
  int j = i & (H_ - 1);
  size_t base = (size_t)b * H3_ + j;
  float ir = GI[base], iz = GI[base + H_], in_ = GI[base + 2 * H_];
  float hr = GH[base], hz = GH[base + H_], hn  = GH[base + 2 * H_];
  float r = 1.0f / (1.0f + __expf(-(ir + hr)));
  float z = 1.0f / (1.0f + __expf(-(iz + hz)));
  float n = tanhf(in_ + r * hn);
  float h = (1.0f - z) * n + z * hprev[i];
  hnew[i]   = h;
  hnew16[i] = (h16)h;
}

// ---------------- output combine: out[b,t,:] = tanh(O1) + tanh(O2) ----------------

__global__ __launch_bounds__(256)
void k_out_combine(const float* __restrict__ O1, const float* __restrict__ O2,
                   float* __restrict__ out, int t) {
  int i = blockIdx.x * blockDim.x + threadIdx.x;   // over B*OUT
  if (i >= B_ * OUT_) return;
  int b = i >> 9;           // OUT_ = 512
  int n = i & (OUT_ - 1);
  out[((size_t)b * T_ + t) * OUT_ + n] = tanhf(O1[i]) + tanhf(O2[i]);
}

// ---------------- aux loss: accum += mean(h2^2) ----------------

__global__ __launch_bounds__(256)
void k_aux_sumsq(const float* __restrict__ h2, float* __restrict__ accum) {
  __shared__ float sm[8];
  float s = 0.0f;
  for (int i = blockIdx.x * blockDim.x + threadIdx.x; i < B_ * H_;
       i += gridDim.x * blockDim.x) {
    float v = h2[i];
    s += v * v;
  }
  for (int off = 16; off > 0; off >>= 1) s += __shfl_down(s, off, 32);
  int lane = threadIdx.x & 31, wave = threadIdx.x >> 5;
  if (lane == 0) sm[wave] = s;
  __syncthreads();
  if (wave == 0) {
    s = (lane < 8) ? sm[lane] : 0.0f;
    for (int off = 4; off > 0; off >>= 1) s += __shfl_down(s, off, 32);
    if (lane == 0) atomicAdd(accum, s * (1.0f / ((float)B_ * (float)H_)));
  }
}

// ---------------- driver ----------------

extern "C" void kernel_launch(void* const* d_in, const int* in_sizes, int n_in,
                              void* d_out, int out_size, void* d_ws, size_t ws_size,
                              hipStream_t stream) {
  const float* x     = (const float*)d_in[0];
  const float* W_ih1 = (const float*)d_in[1];
  const float* W_hh1 = (const float*)d_in[2];
  const float* b_ih1 = (const float*)d_in[3];
  const float* b_hh1 = (const float*)d_in[4];
  const float* W_ih2 = (const float*)d_in[5];
  const float* W_hh2 = (const float*)d_in[6];
  const float* b_ih2 = (const float*)d_in[7];
  const float* b_hh2 = (const float*)d_in[8];
  const float* Wo1   = (const float*)d_in[9];
  const float* bo1   = (const float*)d_in[10];
  const float* Wo2   = (const float*)d_in[11];
  const float* bo2   = (const float*)d_in[12];
  float* out = (float*)d_out;
  float* aux = out + (size_t)B_ * T_ * OUT_;

  char* ws = (char*)d_ws;
  size_t off = 0;
  auto alloc = [&](size_t bytes) -> void* {
    void* p = ws + off;
    off += (bytes + 255) & ~(size_t)255;
    return p;
  };

  h16* w_ih1_16 = (h16*)alloc((size_t)H3_ * IN_ * 2);
  h16* w_hh1_16 = (h16*)alloc((size_t)H3_ * H_  * 2);
  h16* w_ih2_16 = (h16*)alloc((size_t)H3_ * H_  * 2);
  h16* w_hh2_16 = (h16*)alloc((size_t)H3_ * H_  * 2);
  h16* wo1_16   = (h16*)alloc((size_t)OUT_ * H_ * 2);
  h16* wo2_16   = (h16*)alloc((size_t)OUT_ * H_ * 2);
  h16* x16      = (h16*)alloc((size_t)B_ * T_ * IN_ * 2);
  h16* h1h      = (h16*)alloc((size_t)B_ * H_ * 2);
  h16* h2h      = (h16*)alloc((size_t)B_ * H_ * 2);
  float* h1f    = (float*)alloc((size_t)B_ * H_ * 4);
  float* h2f    = (float*)alloc((size_t)B_ * H_ * 4);
  float* GI     = (float*)alloc((size_t)B_ * H3_ * 4);
  float* GH     = (float*)alloc((size_t)B_ * H3_ * 4);
  float* O1     = (float*)alloc((size_t)B_ * OUT_ * 4);
  float* O2     = (float*)alloc((size_t)B_ * OUT_ * 4);
  (void)ws_size;

  auto cg = [](int n) { return dim3((unsigned)((n + 255) / 256)); };

  // One-time (per call) conversions and state init.
  k_cvt_f16<<<cg(H3_ * IN_), 256, 0, stream>>>(W_ih1, w_ih1_16, H3_ * IN_);
  k_cvt_f16<<<cg(H3_ * H_ ), 256, 0, stream>>>(W_hh1, w_hh1_16, H3_ * H_);
  k_cvt_f16<<<cg(H3_ * H_ ), 256, 0, stream>>>(W_ih2, w_ih2_16, H3_ * H_);
  k_cvt_f16<<<cg(H3_ * H_ ), 256, 0, stream>>>(W_hh2, w_hh2_16, H3_ * H_);
  k_cvt_f16<<<cg(OUT_ * H_), 256, 0, stream>>>(Wo1,   wo1_16,   OUT_ * H_);
  k_cvt_f16<<<cg(OUT_ * H_), 256, 0, stream>>>(Wo2,   wo2_16,   OUT_ * H_);
  k_cvt_f16<<<cg(B_ * T_ * IN_), 256, 0, stream>>>(x, x16, B_ * T_ * IN_);
  k_zero_f32<<<cg(B_ * H_), 256, 0, stream>>>(h1f, B_ * H_);
  k_zero_f32<<<cg(B_ * H_), 256, 0, stream>>>(h2f, B_ * H_);
  k_zero_f16<<<cg(B_ * H_), 256, 0, stream>>>(h1h, B_ * H_);
  k_zero_f16<<<cg(B_ * H_), 256, 0, stream>>>(h2h, B_ * H_);
  k_zero_f32<<<1, 256, 0, stream>>>(aux, 1);

  const dim3 blk(256);
  const dim3 grid_g(H3_ / 128, B_ / 128);   // 24 x 32 blocks for gate GEMMs
  const dim3 grid_o(OUT_ / 128, B_ / 128);  //  4 x 32 blocks for output GEMMs

  for (int t = 0; t < T_; ++t) {
    // Layer 1: GI1 = x_t * W_ih1^T + b_ih1 ; GH1 = h1 * W_hh1^T + b_hh1
    k_gemm_wmma<<<grid_g, blk, 0, stream>>>(x16 + (size_t)t * IN_, T_ * IN_,
                                            w_ih1_16, IN_, b_ih1, GI, H3_, IN_);
    k_gemm_wmma<<<grid_g, blk, 0, stream>>>(h1h, H_, w_hh1_16, H_, b_hh1, GH, H3_, H_);
    k_gru_gates<<<cg(B_ * H_), blk, 0, stream>>>(GI, GH, h1f, h1f, h1h);

    // Layer 2: GI2 = h1 * W_ih2^T + b_ih2 ; GH2 = h2 * W_hh2^T + b_hh2
    k_gemm_wmma<<<grid_g, blk, 0, stream>>>(h1h, H_, w_ih2_16, H_, b_ih2, GI, H3_, H_);
    k_gemm_wmma<<<grid_g, blk, 0, stream>>>(h2h, H_, w_hh2_16, H_, b_hh2, GH, H3_, H_);
    k_gru_gates<<<cg(B_ * H_), blk, 0, stream>>>(GI, GH, h2f, h2f, h2h);

    // Output: out = tanh(h1*Wo1^T + bo1) + tanh(h2*Wo2^T + bo2)
    k_gemm_wmma<<<grid_o, blk, 0, stream>>>(h1h, H_, wo1_16, H_, bo1, O1, OUT_, H_);
    k_gemm_wmma<<<grid_o, blk, 0, stream>>>(h2h, H_, wo2_16, H_, bo2, O2, OUT_, H_);
    k_out_combine<<<cg(B_ * OUT_), blk, 0, stream>>>(O1, O2, out, t);

    // aux += mean(h2^2)
    k_aux_sumsq<<<dim3(512), blk, 0, stream>>>(h2f, aux);
  }
}
